// EfficientInteractionBilinear_86629490360978
// MI455X (gfx1250) — matrix operand order repaired and checked by reading, not compiled, not run on hardware
//
#include <hip/hip_runtime.h>
#include <hip/hip_fp16.h>

typedef __attribute__((ext_vector_type(2)))  float    v2f;
typedef __attribute__((ext_vector_type(8)))  float    v8f;
typedef __attribute__((ext_vector_type(8)))  _Float16 v8h;
typedef __attribute__((ext_vector_type(16))) _Float16 v16h;

constexpr int N_EDGES = 50000;
constexpr int KMAX    = 24;
constexpr int N_SPH   = 16;
constexpr int EMB     = 64;    // D
constexpr int EMB_I   = 64;    // I
constexpr int UOUT    = 128;   // O
constexpr long N_TRIP = 600000;

constexpr int QPITCH = 68;     // f32 per Q row (64 + 4 pad -> conflict-free)
constexpr int PPITCH = 1032;   // f16 per P edge row (1024 + 8 pad -> 2064B pitch, banks spread by 4)

// ---------------- kernel 1: zero m2 workspace (f16) ----------------
__global__ void zero_m2_kernel(uint4* __restrict__ p, long n16) {
  long t = (long)blockIdx.x * blockDim.x + threadIdx.x;
  if (t < n16) p[t] = uint4{0u, 0u, 0u, 0u};
}

// ---------------- kernel 2: ragged scatter m (f32) -> m2 (f16) ----------------
__global__ void scatter_m_kernel(const float* __restrict__ m,
                                 const int* __restrict__ idr,
                                 const int* __restrict__ irg,
                                 const int* __restrict__ eoff,
                                 _Float16* __restrict__ m2) {
  long t = (long)blockIdx.x * blockDim.x + threadIdx.x;   // N_TRIP * 8 threads
  if (t >= N_TRIP * 8) return;
  int  part = (int)(t & 7);                               // 8 f16 (16B) per thread
  long trip = t >> 3;
  long e = (long)idr[trip] - (long)eoff[0];
  long g = (long)irg[trip];
  const float* src = m + trip * EMB + part * 8;
  v8h v;
  #pragma unroll
  for (int j = 0; j < 8; ++j) v[j] = (_Float16)src[j];
  *(v8h*)(m2 + (e * KMAX + g) * EMB + part * 8) = v;
}

// ------- kernel 3: pack weight (D,I,O) f32 -> f16 B-fragment order -------
// frag index t = ((ntile*128 + kchunk)*32 + lane)*16 + j
// B element: K = kchunk*32 + (j&7) + 8*(lane>>4) + 16*(j>>3), N = ntile*16 + (lane&15)
// flattened K = i*64 + d  (matches A = P[i_local, d] flattening)
__global__ void pack_weight_kernel(const float* __restrict__ w, _Float16* __restrict__ wf) {
  int t = blockIdx.x * blockDim.x + threadIdx.x;          // 64*64*128 = 524288 threads
  int j    = t & 15;
  int lane = (t >> 4) & 31;
  int kcg  = (t >> 9) & 127;
  int nt   = t >> 16;
  int k = kcg * 32 + ((j & 7) + ((lane >> 4) << 3) + ((j >> 3) << 4));
  int i = k >> 6, d = k & 63;
  int o = nt * 16 + (lane & 15);
  wf[t] = (_Float16)w[((long)d * EMB_I + i) * UOUT + o];
}

// ---------------- kernel 4: fused Q -> P -> out ----------------
__global__ __launch_bounds__(256)
void fused_interaction_kernel(const float* __restrict__ rbf,
                              const float* __restrict__ sph,
                              const _Float16* __restrict__ m2,
                              const v16h* __restrict__ wf,
                              float* __restrict__ out) {
  __shared__ __align__(16) float    Qs[16 * N_SPH * QPITCH]; // 16 edges x 16 s x (64+4) d
  __shared__ __align__(16) _Float16 Ps[16 * PPITCH];         // 16 edges x (1024+8) (i_local,d)

  const int w    = threadIdx.x >> 5;
  const int lane = threadIdx.x & 31;
  const int hl   = lane >> 4;
  const int ln   = lane & 15;
  const long eBase = (long)blockIdx.x * 16;

  v8f acc = {};  // persistent step-4 accumulator: rows = 16 edges, cols = o-tile w

  // ---- Phase Q: Q[e][s][d] = sum_k sph[e][s][k] * m2[e][k][d]  (f32 WMMA, K=24) ----
  for (int ee = 0; ee < 2; ++ee) {
    const int  el = w * 2 + ee;
    const long e  = eBase + el;
    const float*    sphE = sph + e * (N_SPH * KMAX) + (long)ln * KMAX;
    const _Float16* m2E  = m2  + e * (KMAX * EMB);
    for (int dt = 0; dt < 4; ++dt) {
      v8f c = {};
      #pragma unroll
      for (int ks = 0; ks < 6; ++ks) {
        const int k0 = ks * 4 + hl * 2;
        v2f a, b;
        a[0] = sphE[k0];
        a[1] = sphE[k0 + 1];
        b[0] = (float)m2E[(k0    ) * EMB + dt * 16 + ln];
        b[1] = (float)m2E[(k0 + 1) * EMB + dt * 16 + ln];
        c = __builtin_amdgcn_wmma_f32_16x16x4_f32(false, a, false, b, (short)0, c, false, false);
      }
      #pragma unroll
      for (int r = 0; r < 8; ++r)
        Qs[el * (N_SPH * QPITCH) + (r + 8 * hl) * QPITCH + dt * 16 + ln] = c[r];
    }
  }
  __syncthreads();

  for (int ic = 0; ic < 4; ++ic) {
    // ---- Phase C: P_chunk[e][i_local][d] = sum_s rbf[e][ic*16+i_local][s] * Q[e][s][d] ----
    for (int ee = 0; ee < 2; ++ee) {
      const int  el = w * 2 + ee;
      const long e  = eBase + el;
      const float* rbfE = rbf + e * (EMB_I * N_SPH) + (long)(ic * 16 + ln) * N_SPH;
      const float* Qe   = Qs + el * (N_SPH * QPITCH);
      for (int dt = 0; dt < 4; ++dt) {
        v8f c = {};
        #pragma unroll
        for (int ks = 0; ks < 4; ++ks) {
          const int k0 = ks * 4 + hl * 2;
          v2f a, b;
          a[0] = rbfE[k0];
          a[1] = rbfE[k0 + 1];
          b[0] = Qe[(k0    ) * QPITCH + dt * 16 + ln];
          b[1] = Qe[(k0 + 1) * QPITCH + dt * 16 + ln];
          c = __builtin_amdgcn_wmma_f32_16x16x4_f32(false, a, false, b, (short)0, c, false, false);
        }
        #pragma unroll
        for (int r = 0; r < 8; ++r)
          Ps[el * PPITCH + (r + 8 * hl) * EMB + dt * 16 + ln] = (_Float16)c[r];
      }
    }
    __syncthreads();

    // ---- Phase D: acc[edge][o] += P_chunk(16 x 1024) @ Wchunk(1024 x 16)  (f16 WMMA) ----
    {
      const _Float16* Prow = Ps + ln * PPITCH + hl * 8;      // A row = edge, kbase = 8*hl
      const v16h*     wfB  = wf + (long)(w * 128 + ic * 32) * 32 + lane;
      #pragma unroll 4
      for (int kc = 0; kc < 32; ++kc) {
        v8h lo = *(const v8h*)(Prow + kc * 32);              // K = kc*32 + kbase + [0..7]
        v8h hi = *(const v8h*)(Prow + kc * 32 + 16);         // K = kc*32 + kbase + 16 + [0..7]
        v16h a = __builtin_shufflevector(lo, hi, 0,1,2,3,4,5,6,7,8,9,10,11,12,13,14,15);
        v16h b = wfB[kc * 32];
        acc = __builtin_amdgcn_wmma_f32_16x16x32_f16(false, a, false, b, (short)0, acc, false, false);
      }
    }
    __syncthreads();
  }

  // ---- writeout: element r -> edge = r + 8*hl, o = w*16 + ln ----
  float* oB = out + eBase * UOUT + w * 16 + ln;
  #pragma unroll
  for (int r = 0; r < 8; ++r)
    oB[(long)(r + 8 * hl) * UOUT] = acc[r];
}

extern "C" void kernel_launch(void* const* d_in, const int* in_sizes, int n_in,
                              void* d_out, int out_size, void* d_ws, size_t ws_size,
                              hipStream_t stream) {
  (void)in_sizes; (void)n_in; (void)out_size; (void)ws_size;
  const float* rbf  = (const float*)d_in[0];
  const float* sph  = (const float*)d_in[1];
  const float* m    = (const float*)d_in[2];
  const int*   idr  = (const int*)d_in[3];
  const int*   irg  = (const int*)d_in[4];
  const float* wgt  = (const float*)d_in[5];
  const int*   eoff = (const int*)d_in[6];

  // workspace layout: [0, 1MB) packed f16 weight fragments, then m2 (E*24*64 f16)
  _Float16* wf = (_Float16*)d_ws;
  _Float16* m2 = (_Float16*)((char*)d_ws + (1u << 20));

  long n16 = (long)N_EDGES * KMAX * EMB * 2 / 16;   // 16-byte chunks of f16 m2
  zero_m2_kernel<<<(unsigned)((n16 + 255) / 256), 256, 0, stream>>>((uint4*)m2, n16);

  long st = N_TRIP * 8;
  scatter_m_kernel<<<(unsigned)((st + 255) / 256), 256, 0, stream>>>(m, idr, irg, eoff, m2);

  pack_weight_kernel<<<(EMB * EMB_I * UOUT + 255) / 256, 256, 0, stream>>>(wgt, wf);

  fused_interaction_kernel<<<N_EDGES / 16, 256, 0, stream>>>(rbf, sph, m2,
                                                             (const v16h*)wf, (float*)d_out);
}
